// Seq2Seq_4733053960314
// MI455X (gfx1250) — matrix operand order repaired and checked by reading, not compiled, not run on hardware
//
#include <hip/hip_runtime.h>

// ---------------------------------------------------------------------------
// Seq2Seq LSTM (B=32, S=T=64, H=512, E=256, V=32000) for MI455X / gfx1250.
//  * all GEMMs: v_wmma_f32_16x16x32_bf16, 64x64 tiles, 8 waves/WG,
//    double-buffered LDS staging via global_load_async_to_lds_b128 (ASYNCcnt)
//  * fragments: 2x ds_load_b128 per operand (16B-aligned padded LDS)
//  * recurrent scan: Whh LDS-resident (256KB bf16, needs CDNA5's 320KB LDS),
//    8 persistent WGs, c-state in registers, h double-buffered bf16 + grid bar
// ---------------------------------------------------------------------------

typedef __bf16 bf16_t;
typedef __attribute__((ext_vector_type(4)))  __bf16 v4bf;
typedef __attribute__((ext_vector_type(8)))  __bf16 v8bf;
typedef __attribute__((ext_vector_type(16))) __bf16 v16bf;
typedef __attribute__((ext_vector_type(8)))  float  v8f;
typedef __attribute__((ext_vector_type(4)))  int    v4i;

#define SCAN_NWG 8

#if __has_builtin(__builtin_amdgcn_global_load_async_to_lds_b128)
#define HAVE_ASYNC 1
// Builtin expects AS1 (global) / AS3 (LDS) pointers to 128-bit int vectors.
typedef __attribute__((address_space(1))) v4i as1_v4i;
typedef __attribute__((address_space(3))) v4i as3_v4i;
#else
#define HAVE_ASYNC 0
#endif

__device__ __forceinline__ float sigm(float x) { return 1.0f / (1.0f + __expf(-x)); }

__device__ __forceinline__ v8f wmma_bf16(v16bf a, v16bf b, v8f c) {
  return __builtin_amdgcn_wmma_f32_16x16x32_bf16(false, a, false, b, (short)0, c,
                                                 false, false);
}

// 16-byte global -> LDS copy (async path on gfx1250, ASYNCcnt-tracked).
__device__ __forceinline__ void copy16(void* lds, const void* g) {
#if HAVE_ASYNC
  __builtin_amdgcn_global_load_async_to_lds_b128((as1_v4i*)g, (as3_v4i*)lds, 0, 0);
#else
  *(uint4*)lds = *(const uint4*)g;
#endif
}

__device__ __forceinline__ void copy_fence() {
#if HAVE_ASYNC
#if __has_builtin(__builtin_amdgcn_s_wait_asynccnt)
  __builtin_amdgcn_s_wait_asynccnt(0);
#else
  asm volatile("s_wait_asynccnt 0x0" ::: "memory");
#endif
#endif
}

// Build a 16-element bf16 fragment from two 128-bit LDS chunks.
__device__ __forceinline__ v16bf frag2(const bf16_t* p0, const bf16_t* p1) {
  v8bf lo = *(const v8bf*)p0;
  v8bf hi = *(const v8bf*)p1;
  return __builtin_shufflevector(lo, hi, 0, 1, 2, 3, 4, 5, 6, 7, 8, 9, 10, 11, 12,
                                 13, 14, 15);
}

// Multi-WG barrier: one counter slot per step (zeroed before launch).
__device__ __forceinline__ void grid_barrier(int* bar, int slot) {
  __syncthreads();
  if (threadIdx.x == 0) {
    __threadfence();
    __hip_atomic_fetch_add(bar + slot, 1, __ATOMIC_ACQ_REL, __HIP_MEMORY_SCOPE_AGENT);
    while (__hip_atomic_load(bar + slot, __ATOMIC_ACQUIRE, __HIP_MEMORY_SCOPE_AGENT) <
           SCAN_NWG) {
      __builtin_amdgcn_s_sleep(1);
    }
  }
  __syncthreads();
  __threadfence();
}

// ---------------------------------------------------------------------------
// C[m,n] = sum_k X[m,k]*W[n,k] (+bias0[n]+bias1[n]); X,W bf16; out f32.
// Double-buffered async LDS staging: one barrier per K-step, tile k+1 loads
// overlap tile k WMMAs. remapSeq=1: rows m=b*63+t -> out[(b*64+t+1)*N + n].
// ---------------------------------------------------------------------------
__global__ void __launch_bounds__(256)
gemm_bf16_wmma(const bf16_t* __restrict__ X, const bf16_t* __restrict__ W,
               const float* __restrict__ bias0, const float* __restrict__ bias1,
               float* __restrict__ out, int M, int N, int K, int remapSeq) {
  __shared__ alignas(16) bf16_t As[2][64][40];  // 80B rows: aligned, bank-free
  __shared__ alignas(16) bf16_t Ws[2][64][40];

  const int mBlk = blockIdx.x * 64;
  const int nBlk = blockIdx.y * 64;
  const int tid  = threadIdx.x;
  const int lane = tid & 31;
  const int wv   = tid >> 5;
  const int mt   = wv & 3;         // m-tile 0..3
  const int nt0  = (wv >> 2) * 2;  // n-tile pair {0,1} or {2,3}
  const int hi   = lane >> 4;
  const int lr   = lane & 15;

  const int r   = tid >> 2;        // staging: 4 threads/row, 8 bf16 (16B) each
  const int cch = (tid & 3) * 8;
  const bool mOk = (mBlk + r) < M;

  // OOB rows are never overwritten by the async copies: zero them once.
  if (!mOk) {
    *(uint4*)&As[0][r][cch] = uint4{0u, 0u, 0u, 0u};
    *(uint4*)&As[1][r][cch] = uint4{0u, 0u, 0u, 0u};
  }

  const bf16_t* xRow = X + (size_t)(mBlk + r) * K + cch;
  const bf16_t* wRow = W + (size_t)(nBlk + r) * K + cch;

  // Prologue: stage tile 0 into buffer 0.
  if (mOk) copy16(&As[0][r][cch], xRow);
  copy16(&Ws[0][r][cch], wRow);

  v8f acc0 = {};
  v8f acc1 = {};

  int cur = 0;
  for (int k0 = 0; k0 < K; k0 += 32, cur ^= 1) {
    copy_fence();     // tile k resident in LDS
    __syncthreads();  // ... for every wave; prior reads of buf[cur^1] done
    if (k0 + 32 < K) {  // stage tile k+1 into the other buffer (overlapped)
      if (mOk) copy16(&As[cur ^ 1][r][cch], xRow + k0 + 32);
      copy16(&Ws[cur ^ 1][r][cch], wRow + k0 + 32);
      __builtin_prefetch(xRow + k0 + 64, 0, 1);  // global_prefetch_b8
      __builtin_prefetch(wRow + k0 + 64, 0, 1);
    }

    // A frag: lane half hi holds K=hi*8..+8 and K=16+hi*8..+8  (ISA 7.12.2)
    v16bf a =
        frag2(&As[cur][mt * 16 + lr][hi * 8], &As[cur][mt * 16 + lr][16 + hi * 8]);
    // B frag: lane half hi holds K=hi*8..+8 and K=16+hi*8..+8 for column lr
    v16bf b0 = frag2(&Ws[cur][nt0 * 16 + lr][hi * 8],
                     &Ws[cur][nt0 * 16 + lr][16 + hi * 8]);
    v16bf b1 = frag2(&Ws[cur][nt0 * 16 + 16 + lr][hi * 8],
                     &Ws[cur][nt0 * 16 + 16 + lr][16 + hi * 8]);
    acc0 = wmma_bf16(a, b0, acc0);
    acc1 = wmma_bf16(a, b1, acc1);
  }

#pragma unroll
  for (int rr = 0; rr < 8; ++rr) {
    int m = mBlk + mt * 16 + rr + hi * 8;  // C/D layout: lanes 16-31 hold M+8
    if (m >= M) continue;
#pragma unroll
    for (int s = 0; s < 2; ++s) {
      int n = nBlk + (nt0 + s) * 16 + lr;
      float v = (s == 0) ? acc0[rr] : acc1[rr];
      if (bias0) v += bias0[n];
      if (bias1) v += bias1[n];
      size_t o;
      if (remapSeq) {
        int b = m / 63, t = m % 63;
        o = ((size_t)(b * 64 + t + 1)) * (size_t)N + n;
      } else {
        o = (size_t)m * (size_t)N + n;
      }
      out[o] = v;
    }
  }
}

// ---------------------------------------------------------------------------
// Persistent cooperative LSTM scan (see header comment).
// ---------------------------------------------------------------------------
__global__ void __launch_bounds__(256)
lstm_scan(const float* __restrict__ xg,     // [B,T,2048] f32, rows m=b*T+t
          const bf16_t* __restrict__ Whh,   // [2048,512] bf16 (pre-converted)
          const float* __restrict__ h0, const float* __restrict__ c0,  // [32,512]
          bf16_t* __restrict__ ys,          // [B,T,512] bf16
          bf16_t* __restrict__ hdb,         // [2][32][512] bf16 double buffer
          float* __restrict__ hOut, float* __restrict__ cOut,
          int* __restrict__ barrier, int T) {
  extern __shared__ char smem[];
  bf16_t(*whh_s)[520] = (bf16_t(*)[520])smem;             // 256x1040B = 266240B
  bf16_t(*hs)[520]    = (bf16_t(*)[520])(smem + 266240);  // 32x1040B  =  33280B
  float(*gates)[260]  = (float(*)[260])(smem + 266240);   // alias (same bytes)

  const int wg   = blockIdx.x;  // 0..7
  const int tid  = threadIdx.x;
  const int lane = tid & 31;
  const int wv   = tid >> 5;
  const int hi   = lane >> 4;
  const int lr   = lane & 15;

  // Async-fill this WG's 256 Whh rows (row r -> gate*512 + wg*64 + j).
  for (int i = tid; i < 256 * 64; i += 256) {  // 16B chunks
    int rr = i >> 6, cc = (i & 63) * 8;
    int grow = (rr >> 6) * 512 + wg * 64 + (rr & 63);
    copy16(&whh_s[rr][cc], Whh + (size_t)grow * 512 + cc);
  }
  copy_fence();

  // Init c registers (8 owned (b,j) cells per thread) and h double-buffer[0].
  float creg[8];
#pragma unroll
  for (int q = 0; q < 8; ++q) {
    int idx = tid * 8 + q;  // 32 b x 64 j
    int b = idx >> 6, j = idx & 63;
    creg[q] = c0[b * 512 + wg * 64 + j];
    hdb[b * 512 + wg * 64 + j] = (bf16_t)h0[b * 512 + wg * 64 + j];
  }
  grid_barrier(barrier, 0);

  for (int t = 0; t < T; ++t) {
    const bf16_t* hcur = hdb + (size_t)(t & 1) * 32 * 512;
    bf16_t* hnxt = hdb + (size_t)((t + 1) & 1) * 32 * 512;

    for (int i = tid; i < 32 * 64; i += 256) {  // stage h (16B chunks, async)
      int rr = i >> 6, cc = (i & 63) * 8;
      copy16(&hs[rr][cc], hcur + rr * 512 + cc);
    }
    copy_fence();
    __syncthreads();

    // gates(32x256) += h(32x512) * Whh_sliceT(512x256)
    v8f acc00 = {}, acc01 = {}, acc10 = {}, acc11 = {};
#pragma unroll 1
    for (int k0 = 0; k0 < 512; k0 += 32) {
      v16bf a0 = frag2(&hs[lr][k0 + hi * 8], &hs[lr][k0 + 16 + hi * 8]);
      v16bf a1 = frag2(&hs[16 + lr][k0 + hi * 8], &hs[16 + lr][k0 + 16 + hi * 8]);
      v16bf b0 = frag2(&whh_s[wv * 32 + lr][k0 + hi * 8],
                       &whh_s[wv * 32 + lr][k0 + 16 + hi * 8]);
      v16bf b1 = frag2(&whh_s[wv * 32 + 16 + lr][k0 + hi * 8],
                       &whh_s[wv * 32 + 16 + lr][k0 + 16 + hi * 8]);
      acc00 = wmma_bf16(a0, b0, acc00);
      acc01 = wmma_bf16(a0, b1, acc01);
      acc10 = wmma_bf16(a1, b0, acc10);
      acc11 = wmma_bf16(a1, b1, acc11);
    }
    __syncthreads();  // done with hs; alias region becomes `gates`

    // Spill D frags (+ precomputed xg) to LDS in gate-local layout.
    int nl0 = wv * 32 + lr, nl1 = wv * 32 + 16 + lr;
    int gc0 = (nl0 >> 6) * 512 + wg * 64 + (nl0 & 63);
    int gc1 = (nl1 >> 6) * 512 + wg * 64 + (nl1 & 63);
#pragma unroll
    for (int rr = 0; rr < 8; ++rr) {
      int bm0 = rr + hi * 8;       // batch rows 0..15
      int bm1 = 16 + rr + hi * 8;  // batch rows 16..31
      gates[bm0][nl0] = acc00[rr] + xg[((size_t)bm0 * T + t) * 2048 + gc0];
      gates[bm0][nl1] = acc01[rr] + xg[((size_t)bm0 * T + t) * 2048 + gc1];
      gates[bm1][nl0] = acc10[rr] + xg[((size_t)bm1 * T + t) * 2048 + gc0];
      gates[bm1][nl1] = acc11[rr] + xg[((size_t)bm1 * T + t) * 2048 + gc1];
    }
    __syncthreads();

    // Elementwise LSTM cell update for owned (b, j) cells.
#pragma unroll
    for (int q = 0; q < 8; ++q) {
      int idx = tid * 8 + q;
      int b = idx >> 6, j = idx & 63;
      float ig = sigm(gates[b][j]);
      float fg = sigm(gates[b][64 + j]);
      float gg = tanhf(gates[b][128 + j]);
      float og = sigm(gates[b][192 + j]);
      float c = fg * creg[q] + ig * gg;
      float h = og * tanhf(c);
      creg[q] = c;
      ys[((size_t)b * T + t) * 512 + wg * 64 + j] = (bf16_t)h;
      hnxt[b * 512 + wg * 64 + j] = (bf16_t)h;
      if (t == T - 1) {
        hOut[b * 512 + wg * 64 + j] = h;
        cOut[b * 512 + wg * 64 + j] = c;
      }
    }
    grid_barrier(barrier, t + 1);
  }
}

// ---------------------------------------------------------------------------
__global__ void cvt_f32_bf16(const float* __restrict__ src, bf16_t* __restrict__ dst,
                             int n4) {
  int i = blockIdx.x * 256 + threadIdx.x;
  if (i >= n4) return;
  float4 v = ((const float4*)src)[i];
  v4bf o;
  o[0] = (bf16_t)v.x; o[1] = (bf16_t)v.y; o[2] = (bf16_t)v.z; o[3] = (bf16_t)v.w;
  ((v4bf*)dst)[i] = o;
}

__global__ void embed_gather_bf16(const int* __restrict__ ids,
                                  const float* __restrict__ table,
                                  bf16_t* __restrict__ out, int L, int idStride) {
  int i = blockIdx.x * 256 + threadIdx.x;  // over 32*L*64 float4 chunks
  int total = 32 * L * 64;
  if (i >= total) return;
  int e4 = (i & 63) * 4;
  int bt = i >> 6;
  int t = bt % L, b = bt / L;
  int id = ids[b * idStride + t];
  float4 v = *(const float4*)&table[(size_t)id * 256 + e4];
  v4bf o;
  o[0] = (bf16_t)v.x; o[1] = (bf16_t)v.y; o[2] = (bf16_t)v.z; o[3] = (bf16_t)v.w;
  *(v4bf*)&out[(size_t)bt * 256 + e4] = o;
}

__global__ void zero_f32(float* __restrict__ p, int n) {
  int i = blockIdx.x * 256 + threadIdx.x;
  if (i < n) p[i] = 0.0f;
}

__global__ void zero_t0(float* __restrict__ out) {
  int i = blockIdx.x * 256 + threadIdx.x;  // 32*32000
  if (i >= 32 * 32000) return;
  int b = i / 32000, v = i % 32000;
  out[(size_t)b * 64 * 32000 + v] = 0.0f;
}

// ---------------------------------------------------------------------------
extern "C" void kernel_launch(void* const* d_in, const int* in_sizes, int n_in,
                              void* d_out, int out_size, void* d_ws, size_t ws_size,
                              hipStream_t stream) {
  (void)in_sizes; (void)n_in; (void)out_size; (void)ws_size;
  const int*   source   = (const int*)d_in[0];
  const int*   target   = (const int*)d_in[1];
  const float* enc_emb  = (const float*)d_in[2];
  const float* enc_Wih0 = (const float*)d_in[3];
  const float* enc_Whh0 = (const float*)d_in[4];
  const float* enc_bih0 = (const float*)d_in[5];
  const float* enc_bhh0 = (const float*)d_in[6];
  const float* enc_Wih1 = (const float*)d_in[7];
  const float* enc_Whh1 = (const float*)d_in[8];
  const float* enc_bih1 = (const float*)d_in[9];
  const float* enc_bhh1 = (const float*)d_in[10];
  const float* dec_emb  = (const float*)d_in[11];
  const float* dec_Wih0 = (const float*)d_in[12];
  const float* dec_Whh0 = (const float*)d_in[13];
  const float* dec_bih0 = (const float*)d_in[14];
  const float* dec_bhh0 = (const float*)d_in[15];
  const float* dec_Wih1 = (const float*)d_in[16];
  const float* dec_Whh1 = (const float*)d_in[17];
  const float* dec_bih1 = (const float*)d_in[18];
  const float* dec_bhh1 = (const float*)d_in[19];
  const float* aff_W    = (const float*)d_in[20];
  const float* aff_b    = (const float*)d_in[21];
  float* out = (float*)d_out;

  char* ws = (char*)d_ws;
  size_t off = 0;
  auto alloc = [&](size_t bytes) {
    size_t o = off;
    off += (bytes + 255) & ~(size_t)255;
    return o;
  };
  int*    counters = (int*)(ws + alloc(320 * 4));       // 4 scans x 80 slots
  float*  zerosHC  = (float*)(ws + alloc(32 * 512 * 4));
  bf16_t* hdb      = (bf16_t*)(ws + alloc(2 * 32 * 512 * 2));
  const int zeroN  = (int)(off / 4);                    // zero [0, off)
  float*  enc_h0f = (float*)(ws + alloc(32 * 512 * 4));
  float*  enc_c0f = (float*)(ws + alloc(32 * 512 * 4));
  float*  enc_h1f = (float*)(ws + alloc(32 * 512 * 4));
  float*  enc_c1f = (float*)(ws + alloc(32 * 512 * 4));
  float*  dump_h  = (float*)(ws + alloc(32 * 512 * 4));
  float*  dump_c  = (float*)(ws + alloc(32 * 512 * 4));
  bf16_t* embA    = (bf16_t*)(ws + alloc((size_t)32 * 64 * 256 * 2));
  float*  xgA     = (float*)(ws + alloc((size_t)32 * 64 * 2048 * 4));
  float*  xgB     = (float*)(ws + alloc((size_t)32 * 64 * 2048 * 4));
  bf16_t* yA      = (bf16_t*)(ws + alloc((size_t)32 * 64 * 512 * 2));
  bf16_t* yB      = (bf16_t*)(ws + alloc((size_t)32 * 64 * 512 * 2));
  bf16_t* cWih0e  = (bf16_t*)(ws + alloc((size_t)2048 * 256 * 2));
  bf16_t* cWih1e  = (bf16_t*)(ws + alloc((size_t)2048 * 512 * 2));
  bf16_t* cWih0d  = (bf16_t*)(ws + alloc((size_t)2048 * 256 * 2));
  bf16_t* cWih1d  = (bf16_t*)(ws + alloc((size_t)2048 * 512 * 2));
  bf16_t* cWhh0e  = (bf16_t*)(ws + alloc((size_t)2048 * 512 * 2));
  bf16_t* cWhh1e  = (bf16_t*)(ws + alloc((size_t)2048 * 512 * 2));
  bf16_t* cWhh0d  = (bf16_t*)(ws + alloc((size_t)2048 * 512 * 2));
  bf16_t* cWhh1d  = (bf16_t*)(ws + alloc((size_t)2048 * 512 * 2));
  bf16_t* cAffW   = (bf16_t*)(ws + alloc((size_t)32000 * 512 * 2));

  const size_t scanLds = 266240 + 33280;  // 299,520 B < 320 KB/WGP

  // ---- init: zero counters/zerosHC/hdb + logits[:,0,:] ----
  zero_f32<<<(zeroN + 255) / 256, 256, 0, stream>>>((float*)ws, zeroN);
  zero_t0<<<(32 * 32000 + 255) / 256, 256, 0, stream>>>(out);

  // ---- weight pre-conversion to bf16 ----
  auto cvt = [&](const float* s, bf16_t* d, int n) {
    cvt_f32_bf16<<<(n / 4 + 255) / 256, 256, 0, stream>>>(s, d, n / 4);
  };
  cvt(enc_Wih0, cWih0e, 2048 * 256);
  cvt(enc_Wih1, cWih1e, 2048 * 512);
  cvt(dec_Wih0, cWih0d, 2048 * 256);
  cvt(dec_Wih1, cWih1d, 2048 * 512);
  cvt(enc_Whh0, cWhh0e, 2048 * 512);
  cvt(enc_Whh1, cWhh1e, 2048 * 512);
  cvt(dec_Whh0, cWhh0d, 2048 * 512);
  cvt(dec_Whh1, cWhh1d, 2048 * 512);
  cvt(aff_W, cAffW, 32000 * 512);

  // ---- Encoder ----
  embed_gather_bf16<<<(32 * 64 * 64 + 255) / 256, 256, 0, stream>>>(source, enc_emb,
                                                                    embA, 64, 64);
  dim3 gFull(2048 / 64, 2048 / 64);
  gemm_bf16_wmma<<<gFull, 256, 0, stream>>>(embA, cWih0e, enc_bih0, enc_bhh0, xgA,
                                            2048, 2048, 256, 0);
  lstm_scan<<<SCAN_NWG, 256, scanLds, stream>>>(xgA, cWhh0e, zerosHC, zerosHC, yA,
                                                hdb, enc_h0f, enc_c0f, counters + 0,
                                                64);
  gemm_bf16_wmma<<<gFull, 256, 0, stream>>>(yA, cWih1e, enc_bih1, enc_bhh1, xgB,
                                            2048, 2048, 512, 0);
  lstm_scan<<<SCAN_NWG, 256, scanLds, stream>>>(xgB, cWhh1e, zerosHC, zerosHC, yB,
                                                hdb, enc_h1f, enc_c1f, counters + 80,
                                                64);

  // ---- Decoder (teacher-forced, 63 steps) ----
  embed_gather_bf16<<<(32 * 63 * 64 + 255) / 256, 256, 0, stream>>>(target, dec_emb,
                                                                    embA, 63, 64);
  dim3 gDec((2016 + 63) / 64, 2048 / 64);
  gemm_bf16_wmma<<<gDec, 256, 0, stream>>>(embA, cWih0d, dec_bih0, dec_bhh0, xgA,
                                           2016, 2048, 256, 0);
  lstm_scan<<<SCAN_NWG, 256, scanLds, stream>>>(xgA, cWhh0d, enc_h0f, enc_c0f, yA,
                                                hdb, dump_h, dump_c, counters + 160,
                                                63);
  gemm_bf16_wmma<<<gDec, 256, 0, stream>>>(yA, cWih1d, dec_bih1, dec_bhh1, xgB, 2016,
                                           2048, 512, 0);
  lstm_scan<<<SCAN_NWG, 256, scanLds, stream>>>(xgB, cWhh1d, enc_h1f, enc_c1f, yB,
                                                hdb, dump_h, dump_c, counters + 240,
                                                63);

  // ---- Vocab projection: logits[b,t+1,:] = d1[b,t,:] @ aff_W^T + aff_b ----
  dim3 gAff((2016 + 63) / 64, 32000 / 64);
  gemm_bf16_wmma<<<gAff, 256, 0, stream>>>(yB, cAffW, aff_b, nullptr, out, 2016,
                                           32000, 512, 1);
}
